// BasicFlowLayer_21586505630281
// MI455X (gfx1250) — matrix operand order repaired and checked by reading, not compiled, not run on hardware
//
#include <hip/hip_runtime.h>
#include <hip/hip_bf16.h>

// ---------------------------------------------------------------------------
// Modulated deformable-conv alignment block (EDVR/PCD style) for gfx1250.
// All heavy math runs on v_wmma_f32_16x16x32_f16 (f16 in, f32 accumulate).
// LDS tiles are staged with GLOBAL_LOAD_ASYNC_TO_LDS_B128 (ASYNCcnt).
//
//  B=4, H=W=128, NF=64, DG=8, TAPS=9.
//  Pipeline:
//   pack_inputs : nbr/ref f32 NCHW -> xin f16 NHWC[128], nbrh f16 NHWC[64]
//   pack_w x4   : OIHW f32 -> [Cout][tap][Cin] f16 (GEMM A, tap-major K)
//   conv3x3<128,h,lrelu> : xin -> t1
//   conv3x3<64 ,h,lrelu> : t1  -> t2
//   conv3x3<64 ,f32,-  > : t2  -> om  (Cout=216, NHWC f32)
//   deform_wmma : bilinear sample+modulate into LDS, WMMA contract -> d_out
// ---------------------------------------------------------------------------

typedef __attribute__((ext_vector_type(16))) _Float16 v16h;
typedef __attribute__((ext_vector_type(8)))  _Float16 v8h;
typedef __attribute__((ext_vector_type(8)))  float    v8f;
typedef __attribute__((ext_vector_type(4)))  int      v4i;

#define SHUF16(lo, hi) __builtin_shufflevector(lo, hi, 0,1,2,3,4,5,6,7,8,9,10,11,12,13,14,15)

static constexpr int Bn = 4;
static constexpr int Hn = 128;
static constexpr int Wn = 128;
static constexpr int NF = 64;

// ---- async load-to-LDS support (gfx1250) ----------------------------------
#if defined(__has_builtin)
#if __has_builtin(__builtin_amdgcn_global_load_async_to_lds_b128)
#define HAVE_ASYNC_LDS 1
#endif
#endif

#if defined(HAVE_ASYNC_LDS)
typedef __attribute__((address_space(1))) v4i gv4i;   // global v4i
typedef __attribute__((address_space(3))) v4i lv4i;   // LDS v4i
__device__ __forceinline__ void async_b128(const void* g, void* l) {
    __builtin_amdgcn_global_load_async_to_lds_b128((gv4i*)g, (lv4i*)l, 0, 0);
}
#if __has_builtin(__builtin_amdgcn_s_wait_asynccnt)
__device__ __forceinline__ void wait_async0() { __builtin_amdgcn_s_wait_asynccnt(0); }
#else
__device__ __forceinline__ void wait_async0() {
    asm volatile("s_wait_asynccnt 0x0" ::: "memory");
}
#endif
#else
// fallback: synchronous copy through VGPRs
__device__ __forceinline__ void async_b128(const void* g, void* l) {
    *(v8h*)l = *(const v8h*)g;
}
__device__ __forceinline__ void wait_async0() {}
#endif

__device__ __forceinline__ void zero_b128(void* l) {
    v8h z{};
    *(v8h*)l = z;
}

// -------------------------- input packing ---------------------------------
__global__ __launch_bounds__(256)
void pack_inputs(const float* __restrict__ nbr, const float* __restrict__ ref,
                 _Float16* __restrict__ xin, _Float16* __restrict__ nbrh)
{
    size_t i = (size_t)blockIdx.x * 256 + threadIdx.x;   // over B*H*W*128
    int c = (int)(i & 127);
    size_t pix = i >> 7;                                  // b*16384 + y*128 + x
    int b = (int)(pix >> 14);
    size_t hw = pix & 16383;
    float v = (c < NF) ? nbr[((size_t)b * NF + c) * (Hn * Wn) + hw]
                       : ref[((size_t)b * NF + (c - NF)) * (Hn * Wn) + hw];
    _Float16 h = (_Float16)v;
    xin[i] = h;
    if (c < NF) nbrh[pix * NF + c] = h;
}

// ------------------------ weight packing ----------------------------------
// OIHW f32 -> wA[o][tap][c] f16  (GEMM A matrix, K = tap*Cin + c)
__global__ __launch_bounds__(256)
void pack_w(const float* __restrict__ w, _Float16* __restrict__ wA,
            int Cout, int Cin)
{
    int i = blockIdx.x * 256 + threadIdx.x;
    if (i >= Cout * Cin * 9) return;
    int tap = i % 9;
    int t = i / 9;
    int c = t % Cin;
    int o = t / Cin;
    wA[(o * 9 + tap) * Cin + c] = (_Float16)w[i];  // w flat = ((o*Cin+c)*9+tap)
}

// ---------------------- implicit-GEMM 3x3 conv ----------------------------
// Workgroup: 8 waves, one image row (N=128 pixels), M-block = 64 channels.
// Wave grid 4(M) x 2(N); each wave: 1 A frag + 4 B frags + 4 WMMA per K-chunk.
template <int CIN, bool OUTF32, bool LRELU>
__global__ __launch_bounds__(256)
void conv3x3_wmma(const _Float16* __restrict__ xin,  // [B][H][W][CIN]
                  const _Float16* __restrict__ wA,   // [Cout][9*CIN]
                  const float*    __restrict__ bias, // [Cout]
                  _Float16* __restrict__ outh,       // !OUTF32: [B][H][W][Cout]
                  float*    __restrict__ outf,       // OUTF32 : [B][H][W][Cout]
                  int Cout)
{
    constexpr int K   = 9 * CIN;
    constexpr int KCH = K / 32;     // K chunks of 32
    constexpr int CPT = CIN / 32;   // chunks per tap (never straddles a tap)

    __shared__ _Float16 As[64 * 32];    // A chunk  [64 ch][32 k]
    __shared__ _Float16 Bs[128 * 32];   // B^T chunk [128 px][32 k]

    const int row  = blockIdx.x;        // 0..B*H-1
    const int bi   = row >> 7;
    const int y    = row & 127;
    const int mblk = blockIdx.y;
    const int tid  = threadIdx.x;
    const int lane = tid & 31;
    const int wave = tid >> 5;
    const int mi   = wave >> 1;         // 0..3
    const int nj   = wave & 1;          // 0..1
    const int r    = lane & 15;
    const int hh   = lane >> 4;

    v8f acc[4] = {};

    for (int kc = 0; kc < KCH; ++kc) {
        const int tap = kc / CPT;
        const int c0  = (kc % CPT) * 32;
        const int ky  = tap / 3 - 1;
        const int kx  = tap % 3 - 1;

        __syncthreads();                 // previous chunk fully consumed
        {   // stage A: 2048 f16, one b128 per thread
            int ar = tid >> 2;
            int ac = (tid & 3) * 8;
            int orow = mblk * 64 + ar;
            _Float16* dst = &As[ar * 32 + ac];
            if (orow < Cout)
                async_b128(&wA[(size_t)orow * K + kc * 32 + ac], dst);
            else
                zero_b128(dst);
        }
        {   // stage B^T: 4096 f16, two b128 per thread, zero-padded halo
            int p    = tid >> 1;
            int part = (tid & 1) * 16;
            int yy = y + ky;
            int xx = p + kx;
            _Float16* dst = &Bs[p * 32 + part];
            if (yy >= 0 && yy < Hn && xx >= 0 && xx < Wn) {
                const _Float16* src =
                    &xin[(((size_t)(bi * Hn + yy) * Wn + xx) * CIN) + c0 + part];
                async_b128(src,     dst);
                async_b128(src + 8, dst + 8);
            } else {
                zero_b128(dst);
                zero_b128(dst + 8);
            }
        }
        wait_async0();                   // this wave's async transfers landed
        __syncthreads();                 // all waves' tiles visible

        // A fragment: 16-bit A 16x32 layout (two contiguous 8-K runs per lane)
        v8h alo = *(const v8h*)&As[(mi * 16 + r) * 32 + hh * 8];
        v8h ahi = *(const v8h*)&As[(mi * 16 + r) * 32 + 16 + hh * 8];
        v16h af = SHUF16(alo, ahi);

#pragma unroll
        for (int nt = 0; nt < 4; ++nt) {
            // B fragment from B^T: two 8-K runs per lane-half (matches A)
            int brow = nj * 64 + nt * 16 + r;
            v8h blo = *(const v8h*)&Bs[brow * 32 + hh * 8];
            v8h bhi = *(const v8h*)&Bs[brow * 32 + 16 + hh * 8];
            v16h bf = SHUF16(blo, bhi);
            acc[nt] = __builtin_amdgcn_wmma_f32_16x16x32_f16(
                false, af, false, bf, (short)0, acc[nt], false, false);
        }
    }

    // epilogue: C layout -> lane holds (co = cobase+p, x = ncol) elements
    int cobase = mblk * 64 + mi * 16 + hh * 8;
    if (cobase < Cout) {
        float bs[8];
#pragma unroll
        for (int p = 0; p < 8; ++p) bs[p] = bias[cobase + p];
#pragma unroll
        for (int nt = 0; nt < 4; ++nt) {
            int x = nj * 64 + nt * 16 + r;
            size_t pix = (size_t)(bi * Hn + y) * Wn + x;
            if constexpr (OUTF32) {
#pragma unroll
                for (int p = 0; p < 8; ++p) {
                    float v = acc[nt][p] + bs[p];
                    if (LRELU) v = (v >= 0.f) ? v : 0.1f * v;
                    outf[pix * Cout + cobase + p] = v;
                }
            } else {
                v8h o{};
#pragma unroll
                for (int p = 0; p < 8; ++p) {
                    float v = acc[nt][p] + bs[p];
                    if (LRELU) v = (v >= 0.f) ? v : 0.1f * v;
                    o[p] = (_Float16)v;
                }
                *(v8h*)&outh[pix * Cout + cobase] = o;
            }
        }
    }
}

// ------------------- deformable conv: sample + WMMA ------------------------
// Phase 1: build modulated bilinear im2col tile [128 px][K=576] f16 in LDS.
// Phase 2: GEMM  out[64][128px] = wA[64][576] * B[576][128]  via WMMA.
// K ordering: k = tap*64 + g*8 + c  (matches pack_w [o][tap][cin]).
__global__ __launch_bounds__(256)
void deform_wmma(const _Float16* __restrict__ nbrh, // [B][H][W][64]
                 const float*    __restrict__ om,   // [B][H][W][216]
                 const _Float16* __restrict__ wA,   // [64][576]
                 const float*    __restrict__ bias, // [64]
                 float*          __restrict__ out)  // [B][64][H][W]
{
    constexpr int KD   = 576;
    constexpr int KROW = 584;            // +8 f16 pad: conflict-free b128 rows
    __shared__ _Float16 Bt[128 * KROW];  // 149.5 KB
    __shared__ _Float16 As[64 * 32];     // 4 KB (streamed A chunk)

    const int rowid = blockIdx.x;
    const int bi = rowid >> 7;
    const int y  = rowid & 127;
    const int tid = threadIdx.x;

    // ---- phase 1: 128 px * 8 groups * 9 taps sampling tasks ----
    for (int t = tid; t < 128 * 72; t += 256) {
        int p   = t / 72;
        int rr  = t % 72;
        int g   = rr / 9;
        int tap = rr % 9;
        int ky  = tap / 3 - 1;
        int kx  = tap % 3 - 1;

        const float* ob = om + ((size_t)(bi * Hn + y) * Wn + p) * 216;
        float oy = ob[g * 9 + tap];
        float ox = ob[72 + g * 9 + tap];
        float ml = ob[144 + g * 9 + tap];
        float mk = 1.0f / (1.0f + expf(-ml));    // sigmoid modulation

        float sy = (float)(y + ky) + oy;
        float sx = (float)(p + kx) + ox;
        float fy0 = floorf(sy), fx0 = floorf(sx);
        int   y0 = (int)fy0,    x0 = (int)fx0;
        float fy = sy - fy0,    fx = sx - fx0;
        float w00 = (1.f - fy) * (1.f - fx), w01 = (1.f - fy) * fx;
        float w10 = fy * (1.f - fx),         w11 = fy * fx;

        float a8[8] = {};
        auto corner = [&](int yy, int xx, float w) {
            if (yy >= 0 && yy < Hn && xx >= 0 && xx < Wn) {
                v8h c8 = *(const v8h*)
                    &nbrh[(((size_t)(bi * Hn + yy) * Wn + xx) * NF) + g * 8];
#pragma unroll
                for (int i = 0; i < 8; ++i) a8[i] += w * (float)c8[i];
            }
        };
        corner(y0,     x0,     w00);
        corner(y0,     x0 + 1, w01);
        corner(y0 + 1, x0,     w10);
        corner(y0 + 1, x0 + 1, w11);

        v8h o{};
#pragma unroll
        for (int i = 0; i < 8; ++i) o[i] = (_Float16)(a8[i] * mk);
        *(v8h*)&Bt[p * KROW + tap * 64 + g * 8] = o;   // k = tap*64 + g*8 + c
    }
    __syncthreads();

    // ---- phase 2: WMMA GEMM, M=64 N=128 K=576 ----
    const int lane = tid & 31;
    const int wave = tid >> 5;
    const int mi = wave >> 1, nj = wave & 1;
    const int r = lane & 15,  hh = lane >> 4;

    v8f acc[4] = {};
    for (int kc = 0; kc < KD / 32; ++kc) {
        __syncthreads();
        {   // stage A chunk [64][32] via async-to-LDS (always in-bounds)
            int ar = tid >> 2;
            int ac = (tid & 3) * 8;
            async_b128(&wA[(size_t)ar * KD + kc * 32 + ac], &As[ar * 32 + ac]);
        }
        wait_async0();
        __syncthreads();

        v8h alo = *(const v8h*)&As[(mi * 16 + r) * 32 + hh * 8];
        v8h ahi = *(const v8h*)&As[(mi * 16 + r) * 32 + 16 + hh * 8];
        v16h af = SHUF16(alo, ahi);
#pragma unroll
        for (int nt = 0; nt < 4; ++nt) {
            int brow = nj * 64 + nt * 16 + r;
            v8h blo = *(const v8h*)&Bt[brow * KROW + kc * 32 + hh * 8];
            v8h bhi = *(const v8h*)&Bt[brow * KROW + kc * 32 + 16 + hh * 8];
            v16h bf = SHUF16(blo, bhi);
            acc[nt] = __builtin_amdgcn_wmma_f32_16x16x32_f16(
                false, af, false, bf, (short)0, acc[nt], false, false);
        }
    }

    // epilogue: f32 NCHW output
    int cobase = mi * 16 + hh * 8;
    float bs[8];
#pragma unroll
    for (int p = 0; p < 8; ++p) bs[p] = bias[cobase + p];
#pragma unroll
    for (int nt = 0; nt < 4; ++nt) {
        int x = nj * 64 + nt * 16 + r;
#pragma unroll
        for (int p = 0; p < 8; ++p) {
            int co = cobase + p;
            out[(((size_t)bi * NF + co) * Hn + y) * Wn + x] = acc[nt][p] + bs[p];
        }
    }
}

// ---------------------------------------------------------------------------
extern "C" void kernel_launch(void* const* d_in, const int* in_sizes, int n_in,
                              void* d_out, int out_size, void* d_ws, size_t ws_size,
                              hipStream_t stream)
{
    (void)in_sizes; (void)n_in; (void)out_size; (void)ws_size;

    const float* nbr    = (const float*)d_in[0];
    const float* ref    = (const float*)d_in[1];
    const float* w_off1 = (const float*)d_in[2];
    const float* b_off1 = (const float*)d_in[3];
    const float* w_off2 = (const float*)d_in[4];
    const float* b_off2 = (const float*)d_in[5];
    const float* w_om   = (const float*)d_in[6];
    const float* b_om   = (const float*)d_in[7];
    const float* w_dcn  = (const float*)d_in[8];
    const float* b_dcn  = (const float*)d_in[9];
    float* out = (float*)d_out;

    const size_t NPIX = (size_t)Bn * Hn * Wn;          // 65536
    char* ws = (char*)d_ws;
    _Float16* xin   = (_Float16*)ws; ws += NPIX * 128 * sizeof(_Float16);
    _Float16* nbrh  = (_Float16*)ws; ws += NPIX * 64  * sizeof(_Float16);
    _Float16* t1    = (_Float16*)ws; ws += NPIX * 64  * sizeof(_Float16);
    _Float16* t2    = (_Float16*)ws; ws += NPIX * 64  * sizeof(_Float16);
    float*    omb   = (float*)ws;    ws += NPIX * 216 * sizeof(float);
    _Float16* wA1   = (_Float16*)ws; ws += (size_t)64  * 1152 * sizeof(_Float16);
    _Float16* wA2   = (_Float16*)ws; ws += (size_t)64  * 576  * sizeof(_Float16);
    _Float16* wAom  = (_Float16*)ws; ws += (size_t)216 * 576  * sizeof(_Float16);
    _Float16* wAdcn = (_Float16*)ws; ws += (size_t)64  * 576  * sizeof(_Float16);

    pack_inputs<<<(int)(NPIX * 128 / 256), 256, 0, stream>>>(nbr, ref, xin, nbrh);
    pack_w<<<(64 * 128 * 9 + 255) / 256, 256, 0, stream>>>(w_off1, wA1, 64, 128);
    pack_w<<<(64 * 64 * 9 + 255) / 256, 256, 0, stream>>>(w_off2, wA2, 64, 64);
    pack_w<<<(216 * 64 * 9 + 255) / 256, 256, 0, stream>>>(w_om, wAom, 216, 64);
    pack_w<<<(64 * 64 * 9 + 255) / 256, 256, 0, stream>>>(w_dcn, wAdcn, 64, 64);

    conv3x3_wmma<128, false, true ><<<dim3(Bn * Hn, 1), 256, 0, stream>>>(
        xin, wA1, b_off1, t1, nullptr, 64);
    conv3x3_wmma<64,  false, true ><<<dim3(Bn * Hn, 1), 256, 0, stream>>>(
        t1, wA2, b_off2, t2, nullptr, 64);
    conv3x3_wmma<64,  true,  false><<<dim3(Bn * Hn, 4), 256, 0, stream>>>(
        t2, wAom, b_om, nullptr, omb, 216);

    deform_wmma<<<Bn * Hn, 256, 0, stream>>>(nbrh, omb, wAdcn, b_dcn, out);
}
